// MACE_7275674599670
// MI455X (gfx1250) — compile-verified, matrix operand order checked by reading, not run on hardware
//
#include <hip/hip_runtime.h>
#include <hip/hip_bf16.h>
#include <math.h>

#define NNODES 50000
#define NEDGES 800000
#define HDIM   128
#define NB     8
#define CUT    5.0f
#define PI_F   3.14159265358979323846f

typedef __attribute__((ext_vector_type(16))) __bf16 v16bf;
typedef __attribute__((ext_vector_type(8)))  __bf16 v8bf;
typedef __attribute__((ext_vector_type(8)))  float  v8f;

// ---------------------------------------------------------------------------
// Fragment helpers (CDNA5 WMMA bf16 16x16x32 layouts, CDNA5 ISA)
// A 16x32: lane L holds row M=L&15; kb = 8*(L>>4); elems 0..7 -> K=kb..kb+7,
//          elems 8..15 -> K=kb+16..kb+23.
// B 32x16 stored pre-shuffled in global: frag[(kt*NT+nt)*32 + lane][16].
// C/D f32: vgpr i, lane L: row = i + 8*(L>>4), col = L&15.
// ---------------------------------------------------------------------------
__device__ inline v16bf load_a_frag(const __bf16* base, int kt, int stride) {
  int lane = threadIdx.x & 31;
  int m    = lane & 15;
  int kb   = (lane >> 4) * 8;
  const __bf16* p = base + m * stride + kt * 32 + kb;
  v8bf lo = *(const v8bf*)(p);
  v8bf hi = *(const v8bf*)(p + 16);
  v16bf a;
#pragma unroll
  for (int i = 0; i < 8; ++i) { a[i] = lo[i]; a[i + 8] = hi[i]; }
  return a;
}

__device__ inline v16bf load_b_frag(const __bf16* wf, int kt, int nt, int NT) {
  int lane = threadIdx.x & 31;
  return *(const v16bf*)(wf + (((size_t)(kt * NT + nt) * 32 + lane) << 4));
}

__device__ inline v8f wmma_bf16(v16bf a, v16bf b, v8f c) {
  return __builtin_amdgcn_wmma_f32_16x16x32_bf16(false, a, false, b,
                                                 (short)0, c, false, false);
}

// fast silu: single v_rcp_f32 instead of IEEE div chain
__device__ inline float silu(float v) {
  return v * __builtin_amdgcn_rcpf(1.0f + __expf(-v));
}

// ---------------------------------------------------------------------------
// Prep: shuffle f32 weights [K, ncols] into bf16 B-fragment layout.
// colmul==3: sum groups of 3 columns (w3 -> sum over LMAX+1 channels).
// ---------------------------------------------------------------------------
__global__ void prep_bfrag(const float* __restrict__ W, __bf16* __restrict__ dst,
                           int K, int ncols, int colmul, int KT, int NT) {
  int gid = blockIdx.x * blockDim.x + threadIdx.x;
  int total = KT * NT * 32 * 16;
  if (gid >= total) return;
  int i    = gid & 15;
  int lane = (gid >> 4) & 31;
  int tile = gid >> 9;
  int nt = tile % NT, kt = tile / NT;
  int n  = lane & 15;
  int kb = (lane >> 4) * 8;
  int k  = kt * 32 + kb + (i < 8 ? i : i + 8);
  float v = 0.0f;
  if (k < K) {
    int col = nt * 16 + n;
    if (colmul == 1) v = W[(size_t)k * ncols + col];
    else { for (int l = 0; l < 3; ++l) v += W[(size_t)k * ncols + col * 3 + l]; }
  }
  dst[gid] = (__bf16)v;
}

__global__ void prep_b3s(const float* __restrict__ b3, float* __restrict__ b3s) {
  int h = threadIdx.x;
  if (h < HDIM) b3s[h] = b3[h * 3] + b3[h * 3 + 1] + b3[h * 3 + 2];
}

__global__ void gather_feats(const int* __restrict__ z, const float* __restrict__ emb,
                             float* __restrict__ feats, int n) {
  int i = blockIdx.x * blockDim.x + threadIdx.x;
  if (i < n) feats[i] = emb[(size_t)z[i >> 7] * HDIM + (i & 127)];
}

__global__ void zero_f32(float* __restrict__ p, int n) {
  int i = blockIdx.x * blockDim.x + threadIdx.x;
  if (i < n) p[i] = 0.0f;
}

// ---------------------------------------------------------------------------
// Fused edge kernel: rbf -> 3-layer MLP (WMMA bf16) -> msg -> atomic scatter.
// 128 threads (4 waves), 32 edges (two 16-row A tiles) per block.
// Wave w owns output columns [32w, 32w+32); 2x2 register blocking so each
// A/B fragment load feeds two WMMAs.
// ---------------------------------------------------------------------------
__global__ __launch_bounds__(128) void edge_kernel(
    const float* __restrict__ pos, const int* __restrict__ eidx,
    const __bf16* __restrict__ w1f, const float* __restrict__ b1,
    const __bf16* __restrict__ w2f, const float* __restrict__ b2,
    const __bf16* __restrict__ w3f, const float* __restrict__ b3s,
    const float* __restrict__ feats, float* __restrict__ agg, int E) {
  __shared__ __bf16 a_lds[2][16][32];
  __shared__ __bf16 h1_lds[2][16][HDIM];
  __shared__ __bf16 h2_lds[2][16][HDIM];
  __shared__ int   erow[32], ecol[32];
  __shared__ float d_lds[32];

  int t = threadIdx.x, lane = t & 31, wv = t >> 5;
  int ebase = blockIdx.x * 32;

  if (t < 32) {
    int e = ebase + t;
    int r = eidx[e], c = eidx[E + e];
    float dx = pos[3 * c + 0] - pos[3 * r + 0];
    float dy = pos[3 * c + 1] - pos[3 * r + 1];
    float dz = pos[3 * c + 2] - pos[3 * r + 2];
    erow[t] = r; ecol[t] = c;
    d_lds[t] = sqrtf(dx * dx + dy * dy + dz * dz);
  }
  __syncthreads();
#pragma unroll
  for (int it = 0; it < 2; ++it) {
    int i  = t + it * 128;
    int er = i >> 3, nb = i & 7;
    float d   = d_lds[er];
    float env = 0.5f * (__cosf(d * (PI_F / CUT)) + 1.0f) * (d < CUT ? 1.0f : 0.0f);
    float rbf = __sinf(d * (float)(nb + 1) * (PI_F / CUT)) *
                __builtin_amdgcn_rcpf(fmaxf(d, 1e-8f)) * env;
    __bf16* row = &a_lds[er >> 4][er & 15][0];
    row[nb]      = (__bf16)rbf;   // K pad 8..31 with zeros
    row[nb + 8]  = (__bf16)0.0f;
    row[nb + 16] = (__bf16)0.0f;
    row[nb + 24] = (__bf16)0.0f;
  }
  __syncthreads();

  // h1 = silu(rbf @ W1 + b1)   (K=32, one K-tile)
  {
    v16bf a0 = load_a_frag(&a_lds[0][0][0], 0, 32);
    v16bf a1 = load_a_frag(&a_lds[1][0][0], 0, 32);
#pragma unroll
    for (int j = 0; j < 2; ++j) {
      int nt = 2 * wv + j;
      v16bf b = load_b_frag(w1f, 0, nt, 8);
      v8f c0 = {}, c1 = {};
      c0 = wmma_bf16(a0, b, c0);
      c1 = wmma_bf16(a1, b, c1);
      int col = nt * 16 + (lane & 15);
      float bias = b1[col];
#pragma unroll
      for (int i = 0; i < 8; ++i) {
        int r = i + ((lane >> 4) * 8);
        h1_lds[0][r][col] = (__bf16)silu(c0[i] + bias);
        h1_lds[1][r][col] = (__bf16)silu(c1[i] + bias);
      }
    }
  }
  __syncthreads();

  // h2 = silu(h1 @ W2 + b2)    (K=128 -> 4 K-tiles, 2x2 blocking)
  {
    v8f acc[2][2] = {};
#pragma unroll
    for (int kt = 0; kt < 4; ++kt) {
      v16bf a0 = load_a_frag(&h1_lds[0][0][0], kt, HDIM);
      v16bf a1 = load_a_frag(&h1_lds[1][0][0], kt, HDIM);
      v16bf b0 = load_b_frag(w2f, kt, 2 * wv + 0, 8);
      v16bf b1 = load_b_frag(w2f, kt, 2 * wv + 1, 8);
      acc[0][0] = wmma_bf16(a0, b0, acc[0][0]);
      acc[0][1] = wmma_bf16(a0, b1, acc[0][1]);
      acc[1][0] = wmma_bf16(a1, b0, acc[1][0]);
      acc[1][1] = wmma_bf16(a1, b1, acc[1][1]);
    }
#pragma unroll
    for (int s = 0; s < 2; ++s)
#pragma unroll
      for (int j = 0; j < 2; ++j) {
        int col = (2 * wv + j) * 16 + (lane & 15);
        float bias = b2[col];
#pragma unroll
        for (int i = 0; i < 8; ++i) {
          int r = i + ((lane >> 4) * 8);
          h2_lds[s][r][col] = (__bf16)silu(acc[s][j][i] + bias);
        }
      }
  }
  __syncthreads();

  // rwsum = h2 @ W3s + b3s; msg = feats[col]*rwsum; agg[row] += msg
  {
    v8f acc[2][2] = {};
#pragma unroll
    for (int kt = 0; kt < 4; ++kt) {
      v16bf a0 = load_a_frag(&h2_lds[0][0][0], kt, HDIM);
      v16bf a1 = load_a_frag(&h2_lds[1][0][0], kt, HDIM);
      v16bf b0 = load_b_frag(w3f, kt, 2 * wv + 0, 8);
      v16bf b1 = load_b_frag(w3f, kt, 2 * wv + 1, 8);
      acc[0][0] = wmma_bf16(a0, b0, acc[0][0]);
      acc[0][1] = wmma_bf16(a0, b1, acc[0][1]);
      acc[1][0] = wmma_bf16(a1, b0, acc[1][0]);
      acc[1][1] = wmma_bf16(a1, b1, acc[1][1]);
    }
#pragma unroll
    for (int s = 0; s < 2; ++s)
#pragma unroll
      for (int j = 0; j < 2; ++j) {
        int col = (2 * wv + j) * 16 + (lane & 15);
        float bias = b3s[col];
#pragma unroll
        for (int i = 0; i < 8; ++i) {
          int r  = i + ((lane >> 4) * 8);
          int er = s * 16 + r;
          float rwv = acc[s][j][i] + bias;
          float fj  = feats[(size_t)ecol[er] * HDIM + col];
          atomicAdd(&agg[(size_t)erow[er] * HDIM + col], fj * rwv);
        }
      }
  }
}

// ---------------------------------------------------------------------------
// Node update: upd = [feats|agg] @ lin_w + lin_b; x = feats+upd; LayerNorm.
// 128 threads, 32 nodes per block (tail-guarded), K = 256 -> 8 K-tiles,
// 2x2 register blocking.
// ---------------------------------------------------------------------------
__global__ __launch_bounds__(128) void node_kernel(
    const __bf16* __restrict__ linwf, const float* __restrict__ lin_b,
    const float* __restrict__ ln_g, const float* __restrict__ ln_bb,
    float* __restrict__ feats, const float* __restrict__ agg, int N) {
  __shared__ __bf16 a_lds[2][16][256];
  __shared__ float  x_lds[2][16][HDIM];
  __shared__ float  mu_s[32], rs_s[32];
  int t = threadIdx.x, lane = t & 31, wv = t >> 5;
  int nbase = blockIdx.x * 32;

  for (int i = t; i < 32 * 256; i += 128) {
    int r = i >> 8, c = i & 255;
    int n = min(nbase + r, N - 1);
    float v = (c < HDIM) ? feats[(size_t)n * HDIM + c]
                         : agg[(size_t)n * HDIM + (c - HDIM)];
    a_lds[r >> 4][r & 15][c] = (__bf16)v;
  }
  __syncthreads();

  {
    v8f acc[2][2] = {};
#pragma unroll
    for (int kt = 0; kt < 8; ++kt) {
      v16bf a0 = load_a_frag(&a_lds[0][0][0], kt, 256);
      v16bf a1 = load_a_frag(&a_lds[1][0][0], kt, 256);
      v16bf b0 = load_b_frag(linwf, kt, 2 * wv + 0, 8);
      v16bf b1 = load_b_frag(linwf, kt, 2 * wv + 1, 8);
      acc[0][0] = wmma_bf16(a0, b0, acc[0][0]);
      acc[0][1] = wmma_bf16(a0, b1, acc[0][1]);
      acc[1][0] = wmma_bf16(a1, b0, acc[1][0]);
      acc[1][1] = wmma_bf16(a1, b1, acc[1][1]);
    }
#pragma unroll
    for (int s = 0; s < 2; ++s)
#pragma unroll
      for (int j = 0; j < 2; ++j) {
        int col = (2 * wv + j) * 16 + (lane & 15);
        float bias = lin_b[col];
#pragma unroll
        for (int i = 0; i < 8; ++i) {
          int r = i + ((lane >> 4) * 8);
          int n = min(nbase + s * 16 + r, N - 1);
          x_lds[s][r][col] = feats[(size_t)n * HDIM + col] + acc[s][j][i] + bias;
        }
      }
  }
  __syncthreads();

  if (t < 32) {
    const float* xr = &x_lds[t >> 4][t & 15][0];
    float sum = 0.0f;
    for (int k = 0; k < HDIM; ++k) sum += xr[k];
    float mu = sum * (1.0f / HDIM);
    float v = 0.0f;
    for (int k = 0; k < HDIM; ++k) { float d = xr[k] - mu; v += d * d; }
    mu_s[t] = mu;
    rs_s[t] = __builtin_amdgcn_rsqf(v * (1.0f / HDIM) + 1e-5f);
  }
  __syncthreads();

  for (int i = t; i < 32 * HDIM; i += 128) {
    int r = i >> 7, c = i & 127;
    int n = nbase + r;
    if (n < N)
      feats[(size_t)n * HDIM + c] =
          (x_lds[r >> 4][r & 15][c] - mu_s[r]) * rs_s[r] * ln_g[c] + ln_bb[c];
  }
}

// ---------------------------------------------------------------------------
// Readout: a = silu(feats @ ro_w1 + ro_b1); e = a@ro_w2 + ro_b2 + ae[z]; sum.
// ---------------------------------------------------------------------------
__global__ __launch_bounds__(128) void readout_kernel(
    const __bf16* __restrict__ row1f, const float* __restrict__ ro_b1,
    const float* __restrict__ ro_w2, const float* __restrict__ ro_b2,
    const int* __restrict__ zat, const float* __restrict__ ae_emb,
    const float* __restrict__ feats, float* __restrict__ sumbuf, int N) {
  __shared__ __bf16 a_lds[16][HDIM];
  __shared__ float  act[16][HDIM];
  __shared__ float  red[16];
  int t = threadIdx.x, lane = t & 31, wv = t >> 5;
  int nbase = blockIdx.x * 16;

  for (int i = t; i < 16 * HDIM; i += 128) {
    int r = i >> 7, c = i & 127;
    a_lds[r][c] = (__bf16)feats[(size_t)(nbase + r) * HDIM + c];
  }
  __syncthreads();

#pragma unroll
  for (int j = 0; j < 2; ++j) {
    int nt = 2 * wv + j;
    v8f acc = {};
#pragma unroll
    for (int kt = 0; kt < 4; ++kt)
      acc = wmma_bf16(load_a_frag(&a_lds[0][0], kt, HDIM),
                      load_b_frag(row1f, kt, nt, 8), acc);
    int col = nt * 16 + (lane & 15);
    float bias = ro_b1[col];
#pragma unroll
    for (int i = 0; i < 8; ++i) {
      int r = i + ((lane >> 4) * 8);
      act[r][col] = silu(acc[i] + bias);
    }
  }
  __syncthreads();

  if (t < 16) {
    float e = ro_b2[0];
    for (int k = 0; k < HDIM; ++k) e += act[t][k] * ro_w2[k];
    int node = nbase + t;
    red[t] = e + ae_emb[zat[node]];
  }
  __syncthreads();
  if (t == 0) {
    float s = 0.0f;
    for (int k = 0; k < 16; ++k) s += red[k];
    atomicAdd(sumbuf, s);
  }
}

__global__ void finalize_kernel(const float* __restrict__ sumbuf,
                                const float* __restrict__ scale,
                                const float* __restrict__ shift,
                                float* __restrict__ out) {
  if (threadIdx.x == 0 && blockIdx.x == 0) out[0] = sumbuf[0] * scale[0] + shift[0];
}

// ---------------------------------------------------------------------------
extern "C" void kernel_launch(void* const* d_in, const int* in_sizes, int n_in,
                              void* d_out, int out_size, void* d_ws, size_t ws_size,
                              hipStream_t stream) {
  const int*   atomic_numbers = (const int*)d_in[0];
  const float* pos      = (const float*)d_in[1];
  const int*   edge_idx = (const int*)d_in[2];
  const float* node_emb = (const float*)d_in[3];
  const float* rw_w1 = (const float*)d_in[4];   // [2,8,128]
  const float* rw_b1 = (const float*)d_in[5];   // [2,128]
  const float* rw_w2 = (const float*)d_in[6];   // [2,128,128]
  const float* rw_b2 = (const float*)d_in[7];   // [2,128]
  const float* rw_w3 = (const float*)d_in[8];   // [2,128,384]
  const float* rw_b3 = (const float*)d_in[9];   // [2,384]
  const float* lin_w = (const float*)d_in[10];  // [2,256,128]
  const float* lin_b = (const float*)d_in[11];  // [2,128]
  const float* ln_g  = (const float*)d_in[12];  // [2,128]
  const float* ln_b  = (const float*)d_in[13];  // [2,128]
  const float* ro_w1 = (const float*)d_in[14];  // [128,128]
  const float* ro_b1 = (const float*)d_in[15];  // [128]
  const float* ro_w2 = (const float*)d_in[16];  // [128,1] -> [128]
  const float* ro_b2 = (const float*)d_in[17];  // [1]
  const float* ae_emb = (const float*)d_in[18]; // [100,1]
  const float* scale  = (const float*)d_in[19];
  const float* shift  = (const float*)d_in[20];

  // workspace layout
  char* ws = (char*)d_ws;
  size_t off = 0;
  auto walloc = [&](size_t bytes) -> void* {
    void* p = ws + off;
    off = (off + bytes + 255) & ~(size_t)255;
    return p;
  };
  float*  feats = (float*)walloc((size_t)NNODES * HDIM * 4);
  float*  agg   = (float*)walloc((size_t)NNODES * HDIM * 4);
  __bf16* w1f[2];  __bf16* w2f[2];  __bf16* w3f[2];  __bf16* linwf[2];
  float*  b3s[2];
  for (int l = 0; l < 2; ++l) {
    w1f[l]   = (__bf16*)walloc((size_t)1 * 8 * 512 * 2);
    w2f[l]   = (__bf16*)walloc((size_t)4 * 8 * 512 * 2);
    w3f[l]   = (__bf16*)walloc((size_t)4 * 8 * 512 * 2);
    linwf[l] = (__bf16*)walloc((size_t)8 * 8 * 512 * 2);
    b3s[l]   = (float*)walloc(HDIM * 4);
  }
  __bf16* row1f  = (__bf16*)walloc((size_t)4 * 8 * 512 * 2);
  float*  sumbuf = (float*)walloc(256);

  auto launch_prep = [&](const float* W, __bf16* dst, int K, int ncols,
                         int colmul, int KT, int NT) {
    int total = KT * NT * 32 * 16;
    prep_bfrag<<<(total + 127) / 128, 128, 0, stream>>>(W, dst, K, ncols, colmul, KT, NT);
  };
  for (int l = 0; l < 2; ++l) {
    launch_prep(rw_w1 + (size_t)l * 8 * HDIM,    w1f[l],   8,   HDIM, 1, 1, 8);
    launch_prep(rw_w2 + (size_t)l * HDIM * HDIM, w2f[l],   HDIM, HDIM, 1, 4, 8);
    launch_prep(rw_w3 + (size_t)l * HDIM * 384,  w3f[l],   HDIM, 384,  3, 4, 8);
    launch_prep(lin_w + (size_t)l * 256 * HDIM,  linwf[l], 256, HDIM, 1, 8, 8);
    prep_b3s<<<1, 128, 0, stream>>>(rw_b3 + (size_t)l * 384, b3s[l]);
  }
  launch_prep(ro_w1, row1f, HDIM, HDIM, 1, 4, 8);

  gather_feats<<<(NNODES * HDIM + 255) / 256, 256, 0, stream>>>(
      atomic_numbers, node_emb, feats, NNODES * HDIM);

  for (int l = 0; l < 2; ++l) {
    zero_f32<<<(NNODES * HDIM + 255) / 256, 256, 0, stream>>>(agg, NNODES * HDIM);
    edge_kernel<<<NEDGES / 32, 128, 0, stream>>>(
        pos, edge_idx, w1f[l], rw_b1 + l * HDIM, w2f[l], rw_b2 + l * HDIM,
        w3f[l], b3s[l], feats, agg, NEDGES);
    node_kernel<<<(NNODES + 31) / 32, 128, 0, stream>>>(
        linwf[l], lin_b + l * HDIM, ln_g + l * HDIM, ln_b + l * HDIM,
        feats, agg, NNODES);
  }

  zero_f32<<<1, 32, 0, stream>>>(sumbuf, 1);
  readout_kernel<<<NNODES / 16, 128, 0, stream>>>(
      row1f, ro_b1, ro_w2, ro_b2, atomic_numbers, ae_emb, feats, sumbuf, NNODES);
  finalize_kernel<<<1, 32, 0, stream>>>(sumbuf, scale, shift, (float*)d_out);
}
